// GAT_more_layers_13726715478301
// MI455X (gfx1250) — matrix operand order repaired and checked by reading, not compiled, not run on hardware
//
#include <hip/hip_runtime.h>
#include <hip/hip_bf16.h>
#include <math.h>

typedef __attribute__((ext_vector_type(16))) __bf16 v16bf;
typedef __attribute__((ext_vector_type(8)))  float  v8f;

#define DIM 128

// ---------------------------------------------------------------- helpers
__device__ __forceinline__ float waveSum(float v) {
    #pragma unroll
    for (int o = 16; o > 0; o >>= 1) v += __shfl_xor(v, o, 32);
    return v;
}

__device__ __forceinline__ void atomicMaxF(float* a, float v) {
    if (v >= 0.0f) atomicMax((int*)a, __float_as_int(v));
    else           atomicMin((unsigned int*)a, __float_as_uint(v));
}

// ---------------------------------------------------------------- converts
__global__ void k_f32_to_bf16(const float* __restrict__ x, __bf16* __restrict__ o, int n) {
    int i = blockIdx.x * blockDim.x + threadIdx.x;
    if (i < n) o[i] = (__bf16)x[i];
}

// W,[k][n] row-major f32 -> WT [n][k] bf16 (both weight matrices in one pass)
__global__ void k_w_transpose2(const float* __restrict__ W, const float* __restrict__ LW,
                               __bf16* __restrict__ WT, __bf16* __restrict__ LWT) {
    int i = blockIdx.x * blockDim.x + threadIdx.x; // 16384
    int k = i >> 7, n = i & 127;
    WT [n * 128 + k] = (__bf16)W [k * 128 + n];
    LWT[n * 128 + k] = (__bf16)LW[k * 128 + n];
}

// ---------------------------------------------------------------- fused WMMA layer front-end
// Computes, per 16-row strip (one wave):
//   h    = Xb @ W                      -> h buffer
//   Xout = Xb @ lin_W + (b + lin_b)    -> aggregation buffer (edge atomics land on top)
//   alpha_s = h @ a_src, alpha_d = h @ a_dst   (reduced straight out of accumulators)
//   mmax = -inf, denom = 0
union FragBF { uint4 u[2]; v16bf v; };

__global__ __launch_bounds__(128)
void k_gat_front(const __bf16* __restrict__ Xb,
                 const __bf16* __restrict__ WT, const __bf16* __restrict__ LWT,
                 const float* __restrict__ a_src, const float* __restrict__ a_dst,
                 const float* __restrict__ bias,  const float* __restrict__ lbias,
                 float* __restrict__ h, float* __restrict__ Xout,
                 float* __restrict__ alpha_s, float* __restrict__ alpha_d,
                 float* __restrict__ mmax, float* __restrict__ denom, int M) {
    __shared__ __bf16 sW[128 * 128];   // 32 KB  [n][k]
    __shared__ __bf16 sL[128 * 128];   // 32 KB  [n][k]
    {
        const uint4* gw = (const uint4*)WT;
        const uint4* gl = (const uint4*)LWT;
        uint4* s0 = (uint4*)sW;
        uint4* s1 = (uint4*)sL;
        #pragma unroll
        for (int i = threadIdx.x; i < 128 * 128 / 8; i += 128) { s0[i] = gw[i]; s1[i] = gl[i]; }
    }
    __syncthreads();

    const int wave = threadIdx.x >> 5;
    const int lane = threadIdx.x & 31;
    const int rowbase = (blockIdx.x * 4 + wave) * 16;
    if (rowbase >= M) return;

    const int r  = lane & 15;        // row-in-tile (A) / column (B,D)
    const int hi = lane >> 4;        // lane half selector

    // A fragments, ISA 16-bit 16x32 layout:
    // lane<16: K = {ks*32+0..7, ks*32+16..23}; lane>=16: K = {ks*32+8..15, ks*32+24..31}
    FragBF a[4];
    const __bf16* xrow = Xb + (size_t)(rowbase + r) * 128;
    #pragma unroll
    for (int ks = 0; ks < 4; ++ks) {
        const int kb = ks * 32 + hi * 8;
        a[ks].u[0] = *(const uint4*)(xrow + kb);
        a[ks].u[1] = *(const uint4*)(xrow + kb + 16);
    }

    float pS[8], pD[8];
    #pragma unroll
    for (int j = 0; j < 8; ++j) { pS[j] = 0.0f; pD[j] = 0.0f; }

    #pragma unroll
    for (int nt = 0; nt < 8; ++nt) {
        const int col = nt * 16 + r;
        const __bf16* wrow = sW + col * 128 + hi * 8;    // K {0..7,16..23} / {8..15,24..31} of window
        const __bf16* lrow = sL + col * 128 + hi * 8;
        v8f acc  = {};
        v8f accS = {};
        #pragma unroll
        for (int ks = 0; ks < 4; ++ks) {
            FragBF bw, bl;
            const uint4* qw = (const uint4*)(wrow + ks * 32);
            const uint4* ql = (const uint4*)(lrow + ks * 32);
            bw.u[0] = qw[0]; bw.u[1] = qw[2];
            bl.u[0] = ql[0]; bl.u[1] = ql[2];
            acc  = __builtin_amdgcn_wmma_f32_16x16x32_bf16(
                       false, a[ks].v, false, bw.v, (short)0, acc,  false, false);
            accS = __builtin_amdgcn_wmma_f32_16x16x32_bf16(
                       false, a[ks].v, false, bl.v, (short)0, accS, false, false);
        }
        const float asc  = a_src[col];
        const float adc  = a_dst[col];
        const float badd = bias[col] + lbias[col];
        // D layout: VGPR j -> row j + 8*hi, col = lane&15
        #pragma unroll
        for (int j = 0; j < 8; ++j) {
            const int row = rowbase + j + hi * 8;
            const float hv = acc[j];
            h   [(size_t)row * 128 + col] = hv;
            Xout[(size_t)row * 128 + col] = accS[j] + badd;
            pS[j] += hv * asc;
            pD[j] += hv * adc;
        }
    }

    // reduce alpha partials across the 16 lanes (r = 0..15) sharing each row
    #pragma unroll
    for (int j = 0; j < 8; ++j) {
        float s = pS[j], d = pD[j];
        #pragma unroll
        for (int m = 8; m >= 1; m >>= 1) {
            s += __shfl_xor(s, m, 32);
            d += __shfl_xor(d, m, 32);
        }
        if (r == 0) {
            const int row = rowbase + j + hi * 8;
            alpha_s[row] = s;
            alpha_d[row] = d;
            mmax[row]    = -INFINITY;
            denom[row]   = 0.0f;
        }
    }
}

// ---------------------------------------------------------------- edge passes
__global__ void k_edge_max(const long long* __restrict__ src, const long long* __restrict__ dst,
                           const float* __restrict__ as, const float* __restrict__ ad,
                           float* __restrict__ e, float* __restrict__ mmax, int nE) {
    int i = blockIdx.x * blockDim.x + threadIdx.x;
    if (i >= nE) return;
    const int s = (int)src[i], d = (int)dst[i];
    float v = as[s] + ad[d];
    v = (v >= 0.0f) ? v : 0.2f * v;      // leaky relu
    e[i] = v;
    atomicMaxF(&mmax[d], v);
}

__global__ void k_edge_exp(const long long* __restrict__ dst, const float* __restrict__ mmax,
                           float* __restrict__ e, float* __restrict__ denom, int nE) {
    int i = blockIdx.x * blockDim.x + threadIdx.x;
    if (i >= nE) return;
    const int d = (int)dst[i];
    const float ex = expf(e[i] - mmax[d]);
    e[i] = ex;
    atomicAdd(&denom[d], ex);
}

// wave per edge: 128-dim weighted gather + atomic scatter
__global__ void k_edge_agg(const long long* __restrict__ src, const long long* __restrict__ dst,
                           const float* __restrict__ ex, const float* __restrict__ denom,
                           const float* __restrict__ h, float* __restrict__ out, int nE) {
    const int edge = (blockIdx.x * blockDim.x + threadIdx.x) >> 5;
    const int lane = threadIdx.x & 31;
    if (edge >= nE) return;
    const int s = (int)src[edge], d = (int)dst[edge];
    const float w = ex[edge] / (denom[d] + 1e-16f);
    const float4 hv = ((const float4*)(h + (size_t)s * 128))[lane];
    float* op = out + (size_t)d * 128 + lane * 4;
    atomicAdd(op + 0, hv.x * w);
    atomicAdd(op + 1, hv.y * w);
    atomicAdd(op + 2, hv.z * w);
    atomicAdd(op + 3, hv.w * w);
}

__global__ void k_relu(float* __restrict__ x, int n) {
    int i = blockIdx.x * blockDim.x + threadIdx.x;
    if (i < n) x[i] = fmaxf(x[i], 0.0f);
}

// ---------------------------------------------------------------- final layer (128 -> 2), scalar
__global__ void k_node2(const float* __restrict__ x, const float* __restrict__ W2,
                        const float* __restrict__ lW2, const float* __restrict__ as2,
                        const float* __restrict__ ad2, const float* __restrict__ b2,
                        const float* __restrict__ lb2, float* __restrict__ h2,
                        float* __restrict__ as, float* __restrict__ ad,
                        float* __restrict__ mmax, float* __restrict__ denom,
                        float* __restrict__ out, int N) {
    const int node = (blockIdx.x * blockDim.x + threadIdx.x) >> 5;
    const int lane = threadIdx.x & 31;
    if (node >= N) return;
    const float4 xv = ((const float4*)(x + (size_t)node * 128))[lane];
    float h0 = 0, h1 = 0, s0 = 0, s1 = 0;
    const float xs[4] = {xv.x, xv.y, xv.z, xv.w};
    #pragma unroll
    for (int t = 0; t < 4; ++t) {
        const int k = lane * 4 + t;
        h0 += xs[t] * W2[k * 2 + 0];  h1 += xs[t] * W2[k * 2 + 1];
        s0 += xs[t] * lW2[k * 2 + 0]; s1 += xs[t] * lW2[k * 2 + 1];
    }
    h0 = waveSum(h0); h1 = waveSum(h1); s0 = waveSum(s0); s1 = waveSum(s1);
    if (lane == 0) {
        h2[node * 2 + 0] = h0; h2[node * 2 + 1] = h1;
        as[node] = h0 * as2[0] + h1 * as2[1];
        ad[node] = h0 * ad2[0] + h1 * ad2[1];
        mmax[node]  = -INFINITY;
        denom[node] = 0.0f;
        out[node * 2 + 0] = s0 + b2[0] + lb2[0];
        out[node * 2 + 1] = s1 + b2[1] + lb2[1];
    }
}

__global__ void k_edge_agg2(const long long* __restrict__ src, const long long* __restrict__ dst,
                            const float* __restrict__ ex, const float* __restrict__ denom,
                            const float* __restrict__ h2, float* __restrict__ out, int nE) {
    int i = blockIdx.x * blockDim.x + threadIdx.x;
    if (i >= nE) return;
    const int s = (int)src[i], d = (int)dst[i];
    const float w = ex[i] / (denom[d] + 1e-16f);
    atomicAdd(&out[d * 2 + 0], w * h2[s * 2 + 0]);
    atomicAdd(&out[d * 2 + 1], w * h2[s * 2 + 1]);
}

// ---------------------------------------------------------------- launcher
extern "C" void kernel_launch(void* const* d_in, const int* in_sizes, int n_in,
                              void* d_out, int out_size, void* d_ws, size_t ws_size,
                              hipStream_t stream) {
    const int N  = in_sizes[0] / DIM;     // 50000
    const int NE = in_sizes[1] / 2;       // 800000

    const float*     x0  = (const float*)d_in[0];
    const long long* ei  = (const long long*)d_in[1];
    const long long* src = ei;
    const long long* dst = ei + NE;

    // bump allocator over workspace (256-byte aligned)
    char* wp = (char*)d_ws;
    auto alloc = [&](size_t bytes) -> char* {
        char* p = wp;
        wp += (bytes + 255) & ~(size_t)255;
        return p;
    };
    float*  h     = (float*)alloc((size_t)N * DIM * 4);
    float*  XA    = (float*)alloc((size_t)N * DIM * 4);
    float*  XB    = (float*)alloc((size_t)N * DIM * 4);
    __bf16* xb    = (__bf16*)alloc((size_t)N * DIM * 2);
    __bf16* wT    = (__bf16*)alloc((size_t)DIM * DIM * 2);
    __bf16* wT2   = (__bf16*)alloc((size_t)DIM * DIM * 2);
    float*  as    = (float*)alloc((size_t)N * 4);
    float*  ad    = (float*)alloc((size_t)N * 4);
    float*  mmax  = (float*)alloc((size_t)N * 4);
    float*  denom = (float*)alloc((size_t)N * 4);
    float*  ebuf  = (float*)alloc((size_t)NE * 4);

    const int elemBlocks = (N * DIM + 255) / 256;
    const int nodeWaveBl = (N * 32 + 255) / 256;
    const int edgeBlocks = (NE + 255) / 256;
    const int edgeWaveBl = (NE * 32 + 255) / 256;
    const int gemmBlocks = (N + 63) / 64;

    const float* xin = x0;
    float* xouts[2] = {XA, XB};

    for (int l = 0; l < 2; ++l) {
        const int pi = 2 + l * 6;
        const float* W     = (const float*)d_in[pi + 0];
        const float* a_src = (const float*)d_in[pi + 1];
        const float* a_dst = (const float*)d_in[pi + 2];
        const float* b     = (const float*)d_in[pi + 3];
        const float* lin_W = (const float*)d_in[pi + 4];
        const float* lin_b = (const float*)d_in[pi + 5];
        float* Xout = xouts[l];

        k_f32_to_bf16<<<elemBlocks, 256, 0, stream>>>(xin, xb, N * DIM);
        k_w_transpose2<<<64, 256, 0, stream>>>(W, lin_W, wT, wT2);
        k_gat_front<<<gemmBlocks, 128, 0, stream>>>(xb, wT, wT2, a_src, a_dst, b, lin_b,
                                                    h, Xout, as, ad, mmax, denom, N);
        k_edge_max<<<edgeBlocks, 256, 0, stream>>>(src, dst, as, ad, ebuf, mmax, NE);
        k_edge_exp<<<edgeBlocks, 256, 0, stream>>>(dst, mmax, ebuf, denom, NE);
        k_edge_agg<<<edgeWaveBl, 256, 0, stream>>>(src, dst, ebuf, denom, h, Xout, NE);
        k_relu<<<elemBlocks, 256, 0, stream>>>(Xout, N * DIM);
        xin = Xout;
    }

    // final layer: 128 -> 2
    {
        const int pi = 2 + 2 * 6;
        const float* W2  = (const float*)d_in[pi + 0];
        const float* as2 = (const float*)d_in[pi + 1];
        const float* ad2 = (const float*)d_in[pi + 2];
        const float* b2  = (const float*)d_in[pi + 3];
        const float* lW2 = (const float*)d_in[pi + 4];
        const float* lb2 = (const float*)d_in[pi + 5];
        float* out = (float*)d_out;
        float* h2  = h;   // reuse, only N*2 used

        k_node2<<<nodeWaveBl, 256, 0, stream>>>(xin, W2, lW2, as2, ad2, b2, lb2,
                                                h2, as, ad, mmax, denom, out, N);
        k_edge_max<<<edgeBlocks, 256, 0, stream>>>(src, dst, as, ad, ebuf, mmax, NE);
        k_edge_exp<<<edgeBlocks, 256, 0, stream>>>(dst, mmax, ebuf, denom, NE);
        k_edge_agg2<<<edgeBlocks, 256, 0, stream>>>(src, dst, ebuf, denom, h2, out, NE);
    }
}
